// PointerNet_2181843386786
// MI455X (gfx1250) — matrix operand order repaired and checked by reading, not compiled, not run on hardware
//
#include <hip/hip_runtime.h>
#include <hip/hip_bf16.h>
#include <stdint.h>

// Problem dims
#define BATCH 512
#define SEQ   256
#define HID   256
#define GATES 1024   // 4*HID

typedef __attribute__((ext_vector_type(16))) _Float16 v16h;
typedef __attribute__((ext_vector_type(8)))  _Float16 v8h;
typedef __attribute__((ext_vector_type(8)))  float    v8f;

// ---------------- workspace layout (bytes) ----------------
#define OFF_WENC    ((size_t)0)          // f16 [1024,256]           524288
#define OFF_WDEC    ((size_t)524288)     // f16 [1024,512]          1048576
#define OFF_EBIAS   ((size_t)1572864)    // f32 [1024]                 4096
#define OFF_DBIAS   ((size_t)1576960)    // f32 [1024]                 4096
#define OFF_HCAT0   ((size_t)1581056)    // f16 [512,512]            524288
#define OFF_HCAT1   ((size_t)2105344)    // f16 [512,512]            524288
#define OFF_C       ((size_t)2629632)    // f32 [512,256]            524288
#define OFF_MASK    ((size_t)3153920)    // i32 [512,256]            524288
#define OFF_ENCOUT  ((size_t)3678208)    // f16 [512,256,256]      67108864
// total ~ 70.8 MB  (enc_out stays resident in MI455X's 192MB L2)

// ---------------- WMMA fragment loaders (16x16x32 f16) ----------------
// A (16x32, M x K), row-major source, lda in halfs.
// lanes 0-15: rows M=0..15, K chunks [k0,k0+8) and [k0+16,k0+24)
// lanes 16-31: rows M=0..15, K chunks [k0+8,k0+16) and [k0+24,k0+32)
__device__ __forceinline__ v16h load_a_frag(const _Float16* __restrict__ A, int lda,
                                            int row, int k0, int lane) {
  const int kb = k0 + ((lane >> 4) & 1) * 8;
  const _Float16* p = A + (size_t)row * lda + kb;
  v8h lo = *(const v8h*)(p);
  v8h hi = *(const v8h*)(p + 16);
  v16h a;
#pragma unroll
  for (int i = 0; i < 8; ++i) { a[i] = lo[i]; a[i + 8] = hi[i]; }
  return a;
}

// B (32x16, K x N) sourced from W row-major [N,K] (i.e. B^T), ldw in halfs.
// lane = column N; lanes 0-15 hold K=[k0,k0+8),[k0+16,k0+24); lanes 16-31 K=[k0+8,k0+16),[k0+24,k0+32).
__device__ __forceinline__ v16h load_b_frag(const _Float16* __restrict__ W, int ldw,
                                            int wrow, int k0, int lane) {
  const int kb = k0 + ((lane >> 4) & 1) * 8;
  const _Float16* p = W + (size_t)wrow * ldw + kb;
  v8h lo = *(const v8h*)(p);
  v8h hi = *(const v8h*)(p + 16);
  v16h b;
#pragma unroll
  for (int i = 0; i < 8; ++i) { b[i] = lo[i]; b[i + 8] = hi[i]; }
  return b;
}

__device__ __forceinline__ float sigmoidf_(float x) { return 1.0f / (1.0f + __expf(-x)); }

__device__ __forceinline__ unsigned hash32(unsigned x) {
  x ^= x >> 16; x *= 0x7feb352dU; x ^= x >> 15; x *= 0x846ca68bU; x ^= x >> 16;
  return x;
}

// ---------------- prep kernels ----------------
__global__ void prep_weights_kernel(const float* __restrict__ encWhh,
                                    const float* __restrict__ decWih,
                                    const float* __restrict__ decWhh,
                                    const float* __restrict__ eb1, const float* __restrict__ eb2,
                                    const float* __restrict__ db1, const float* __restrict__ db2,
                                    _Float16* __restrict__ Wenc, _Float16* __restrict__ Wdec,
                                    float* __restrict__ encB, float* __restrict__ decB) {
  int i = blockIdx.x * blockDim.x + threadIdx.x;         // up to 1024*512
  if (i < GATES * HID) Wenc[i] = (_Float16)encWhh[i];
  if (i < GATES * 2 * HID) {
    int j = i >> 9, k = i & 511;
    float v = (k < HID) ? decWih[j * HID + k] : decWhh[j * HID + (k - HID)];
    Wdec[i] = (_Float16)v;
  }
  if (i < GATES) { encB[i] = eb1[i] + eb2[i]; decB[i] = db1[i] + db2[i]; }
}

__global__ void prep_state_kernel(_Float16* __restrict__ hcat0, _Float16* __restrict__ hcat1,
                                  float* __restrict__ c, int* __restrict__ mask) {
  int i = blockIdx.x * blockDim.x + threadIdx.x;          // up to 512*512
  if (i < BATCH * 2 * HID) { hcat0[i] = (_Float16)0.0f; hcat1[i] = (_Float16)0.0f; }
  if (i < BATCH * HID)     { c[i] = 0.0f; mask[i] = 0; }
}

// ---------------- encoder LSTM step ----------------
// One wave computes a 16(batch) x 16(hidden) tile for ALL FOUR gates, then
// applies the LSTM cell in registers. 512 waves = 128 blocks x 4 waves.
__global__ __launch_bounds__(128) void enc_step_kernel(
    const _Float16* __restrict__ hcat_in,   // [512,512], h at col 256
    _Float16* __restrict__ hcat_out,        // [512,512], write h part
    const _Float16* __restrict__ Wenc,      // [1024,256] f16
    const float* __restrict__ Wih,          // [1024,2]  f32
    const float* __restrict__ bias,         // [1024]    b_ih+b_hh
    const float* __restrict__ x,            // [512,256,2]
    float* __restrict__ c,                  // [512,256]
    _Float16* __restrict__ enc_out,         // [512,256,256]
    int t) {
  const int lane = threadIdx.x & 31;
  const int wid  = blockIdx.x * 4 + (threadIdx.x >> 5);   // 0..511
  const int m0 = (wid & 31) * 16;                         // batch tile
  const int n0 = (wid >> 5) * 16;                         // hidden tile

  v8f acc[4];
#pragma unroll
  for (int g = 0; g < 4; ++g)
#pragma unroll
    for (int r = 0; r < 8; ++r) acc[g][r] = 0.0f;

  const int arow = m0 + (lane & 15);
  const int bcol = n0 + (lane & 15);
#pragma unroll
  for (int k0 = 0; k0 < HID; k0 += 32) {
    v16h a = load_a_frag(hcat_in + HID, 2 * HID, arow, k0, lane);
#pragma unroll
    for (int g = 0; g < 4; ++g) {
      v16h b = load_b_frag(Wenc, HID, g * HID + bcol, k0, lane);
      acc[g] = __builtin_amdgcn_wmma_f32_16x16x32_f16(
          false, a, false, b, (short)0, acc[g], false, false);
    }
  }

  // epilogue: fuse x @ W_ih^T + biases, then LSTM cell
  const int n = n0 + (lane & 15);
  float w0[4], w1[4], bs[4];
#pragma unroll
  for (int g = 0; g < 4; ++g) {
    int rr = g * HID + n;
    w0[g] = Wih[2 * rr]; w1[g] = Wih[2 * rr + 1]; bs[g] = bias[rr];
  }
  const int mbase = m0 + ((lane >> 4) & 1) * 8;
#pragma unroll
  for (int r = 0; r < 8; ++r) {
    int b = mbase + r;
    float x0 = x[(size_t)b * (SEQ * 2) + t * 2 + 0];
    float x1 = x[(size_t)b * (SEQ * 2) + t * 2 + 1];
    float gi = acc[0][r] + x0 * w0[0] + x1 * w1[0] + bs[0];
    float gf = acc[1][r] + x0 * w0[1] + x1 * w1[1] + bs[1];
    float gg = acc[2][r] + x0 * w0[2] + x1 * w1[2] + bs[2];
    float go = acc[3][r] + x0 * w0[3] + x1 * w1[3] + bs[3];
    float iv = sigmoidf_(gi), fv = sigmoidf_(gf), gv = tanhf(gg), ov = sigmoidf_(go);
    size_t ci = (size_t)b * HID + n;
    float cn = fv * c[ci] + iv * gv;
    c[ci] = cn;
    float hn = ov * tanhf(cn);
    _Float16 hh = (_Float16)hn;
    hcat_out[(size_t)b * (2 * HID) + HID + n] = hh;
    enc_out[((size_t)b * SEQ + t) * HID + n]  = hh;
  }
}

// ---------------- decoder LSTM step (fused [dec_in|h] GEMM, K=512) ----------------
__global__ __launch_bounds__(128) void dec_step_kernel(
    const _Float16* __restrict__ hcat_in,   // [512,512] = [dec_in | h]
    _Float16* __restrict__ hcat_out,        // write new h at col 256
    const _Float16* __restrict__ Wdec,      // [1024,512] = [W_ih | W_hh]
    const float* __restrict__ bias,         // [1024]
    float* __restrict__ c) {
  const int lane = threadIdx.x & 31;
  const int wid  = blockIdx.x * 4 + (threadIdx.x >> 5);
  const int m0 = (wid & 31) * 16;
  const int n0 = (wid >> 5) * 16;

  v8f acc[4];
#pragma unroll
  for (int g = 0; g < 4; ++g)
#pragma unroll
    for (int r = 0; r < 8; ++r) acc[g][r] = 0.0f;

  const int arow = m0 + (lane & 15);
  const int bcol = n0 + (lane & 15);
#pragma unroll 4
  for (int k0 = 0; k0 < 2 * HID; k0 += 32) {
    v16h a = load_a_frag(hcat_in, 2 * HID, arow, k0, lane);
#pragma unroll
    for (int g = 0; g < 4; ++g) {
      v16h b = load_b_frag(Wdec, 2 * HID, g * HID + bcol, k0, lane);
      acc[g] = __builtin_amdgcn_wmma_f32_16x16x32_f16(
          false, a, false, b, (short)0, acc[g], false, false);
    }
  }

  const int n = n0 + (lane & 15);
  float bs[4];
#pragma unroll
  for (int g = 0; g < 4; ++g) bs[g] = bias[g * HID + n];
  const int mbase = m0 + ((lane >> 4) & 1) * 8;
#pragma unroll
  for (int r = 0; r < 8; ++r) {
    int b = mbase + r;
    float iv = sigmoidf_(acc[0][r] + bs[0]);
    float fv = sigmoidf_(acc[1][r] + bs[1]);
    float gv = tanhf(acc[2][r] + bs[2]);
    float ov = sigmoidf_(acc[3][r] + bs[3]);
    size_t ci = (size_t)b * HID + n;
    float cn = fv * c[ci] + iv * gv;
    c[ci] = cn;
    float hn = ov * tanhf(cn);
    hcat_out[(size_t)b * (2 * HID) + HID + n] = (_Float16)hn;
  }
}

// ---------------- attention + Gumbel-max sampling + gather ----------------
// One block per batch row; 256 threads, one per sequence position.
__global__ __launch_bounds__(256) void attn_sample_kernel(
    const _Float16* __restrict__ enc_out,   // [512,256,256] (L2-resident)
    _Float16* __restrict__ hcat_cur,        // this step's output buffer: h at +256, dec_in at +0
    int* __restrict__ mask,                 // [512,256]
    int* __restrict__ out_idx,              // [512,256]  (indices, [B,S] layout)
    int t) {
  const int b = blockIdx.x;
  const int s = threadIdx.x;
  __shared__ float hsh[HID];
  __shared__ float sval[SEQ];
  __shared__ int   sidx[SEQ];
  __shared__ int   best;

  hsh[s] = (float)hcat_cur[(size_t)b * (2 * HID) + HID + s];
  __syncthreads();

  float score;
  if (mask[b * SEQ + s]) {
    score = -INFINITY;
  } else {
    float acc = 0.0f;
    const _Float16* e = enc_out + ((size_t)b * SEQ + s) * HID;
#pragma unroll 4
    for (int k = 0; k < HID; k += 8) {
      v8h ev = *(const v8h*)(e + k);
#pragma unroll
      for (int j = 0; j < 8; ++j) acc += (float)ev[j] * hsh[k + j];
    }
    unsigned hsd = hash32((unsigned)(b * SEQ + s) ^ hash32((unsigned)t * 0x9e3779b9U + 0x42u));
    float u = fmaxf((float)(hsd >> 8) * (1.0f / 16777216.0f), 1e-7f);
    float gmb = -__logf(-__logf(u));          // Gumbel(0,1)
    score = acc + gmb;
  }
  sval[s] = score; sidx[s] = s;
  __syncthreads();
#pragma unroll
  for (int st = SEQ / 2; st > 0; st >>= 1) {
    if (s < st) {
      if (sval[s + st] > sval[s]) { sval[s] = sval[s + st]; sidx[s] = sidx[s + st]; }
    }
    __syncthreads();
  }
  if (s == 0) {
    best = sidx[0];
    out_idx[b * SEQ + t] = best;
    mask[b * SEQ + best] = 1;
  }
  __syncthreads();
  // dec_in for next step = enc_out[b, best, :]
  hcat_cur[(size_t)b * (2 * HID) + s] =
      enc_out[((size_t)b * SEQ + best) * HID + s];
}

// ---------------- host launch ----------------
extern "C" void kernel_launch(void* const* d_in, const int* in_sizes, int n_in,
                              void* d_out, int out_size, void* d_ws, size_t ws_size,
                              hipStream_t stream) {
  (void)in_sizes; (void)n_in; (void)out_size; (void)ws_size;
  const float* x       = (const float*)d_in[0];   // [512,256,2]
  const float* encWih  = (const float*)d_in[1];   // [1024,2]
  const float* encWhh  = (const float*)d_in[2];   // [1024,256]
  const float* encBih  = (const float*)d_in[3];
  const float* encBhh  = (const float*)d_in[4];
  const float* decWih  = (const float*)d_in[5];   // [1024,256]
  const float* decWhh  = (const float*)d_in[6];   // [1024,256]
  const float* decBih  = (const float*)d_in[7];
  const float* decBhh  = (const float*)d_in[8];

  char* ws = (char*)d_ws;
  _Float16* Wenc   = (_Float16*)(ws + OFF_WENC);
  _Float16* Wdec   = (_Float16*)(ws + OFF_WDEC);
  float*    encB   = (float*)(ws + OFF_EBIAS);
  float*    decB   = (float*)(ws + OFF_DBIAS);
  _Float16* hcat0  = (_Float16*)(ws + OFF_HCAT0);
  _Float16* hcat1  = (_Float16*)(ws + OFF_HCAT1);
  float*    cbuf   = (float*)(ws + OFF_C);
  int*      mask   = (int*)(ws + OFF_MASK);
  _Float16* encOut = (_Float16*)(ws + OFF_ENCOUT);
  int*      outIdx = (int*)d_out;

  // prep: f16 weights, fused biases, zero state
  prep_weights_kernel<<<(GATES * 2 * HID + 255) / 256, 256, 0, stream>>>(
      encWhh, decWih, decWhh, encBih, encBhh, decBih, decBhh, Wenc, Wdec, encB, decB);
  prep_state_kernel<<<(BATCH * 2 * HID + 255) / 256, 256, 0, stream>>>(
      hcat0, hcat1, cbuf, mask);

  _Float16* hc[2] = {hcat0, hcat1};

  // encoder: 256 sequential steps, ping-pong h buffers
  for (int t = 0; t < SEQ; ++t) {
    enc_step_kernel<<<128, 128, 0, stream>>>(
        hc[t & 1], hc[(t + 1) & 1], Wenc, encWih, encB, x, cbuf, encOut, t);
  }
  // decoder: after t=255 the final h lives in hc[0]; dec_in regions are zero.
  for (int t = 0; t < SEQ; ++t) {
    dec_step_kernel<<<128, 128, 0, stream>>>(
        hc[t & 1], hc[(t + 1) & 1], Wdec, decB, cbuf);
    attn_sample_kernel<<<BATCH, SEQ, 0, stream>>>(
        encOut, hc[(t + 1) & 1], mask, outIdx, t);
  }
}
